// Recurrent_44178033606835
// MI455X (gfx1250) — matrix operand order repaired and checked by reading, not compiled, not run on hardware
//
#include <hip/hip_runtime.h>

#define T_STEPS 1024
#define BATCH   256
#define NH      256   // hidden size
#define NIN     64
#define NOUT    32
#define ALPHA   0.1f

#define KTOT    (NH + NIN)   // 320: combined K = [r | u]
#define NCHUNK  (KTOT / 32)  // 10 WMMA k-chunks (8 over r, 2 over u)

typedef _Float16 v16h __attribute__((ext_vector_type(16)));
typedef _Float16 v8h  __attribute__((ext_vector_type(8)));
typedef float    v8f  __attribute__((ext_vector_type(8)));

// ---------------------------------------------------------------------------
// Recurrent scan: 16 workgroups x 256 threads (8 waves). Each WG owns 16 batch
// rows for all T steps; wave w owns two 16-col N-tiles (cols [w*32, w*32+32)).
//  - Wrec/Win B-fragments resident in VGPRs for the whole kernel (160 VGPRs)
//  - x state resident in registers (16 per lane)
//  - r=relu(x) and u_t double-buffered in LDS  -> ONE barrier per step
//  - noise double-buffered in registers        -> no global latency on path
// ---------------------------------------------------------------------------
__global__ __launch_bounds__(256) void rnn_scan_kernel(
    const float* __restrict__ inputs,   // (T,B,NIN)
    const float* __restrict__ noise,    // (T,B,NH)
    const float* __restrict__ x0,       // (B,NH)
    const float* __restrict__ Win,      // (NIN,NH)
    const float* __restrict__ Wrec,     // (NH,NH)
    const float* __restrict__ brec,     // (NH)
    float* __restrict__ x_out)          // (T,B,NH)
{
    __shared__ _Float16 r2[2][16 * NH];   // relu(x) f16, double-buffered (16 KB)
    __shared__ _Float16 u2[2][16 * NIN];  // u_t f16,     double-buffered (4 KB)

    const int tid   = threadIdx.x;
    const int lane  = tid & 31;
    const int wave  = tid >> 5;           // 0..7
    const int b0    = blockIdx.x * 16;    // batch rows [b0, b0+16)
    const int lhalf = lane >> 4;          // 0 | 1
    const int l16   = lane & 15;
    const int nc0   = wave * 32 + l16;    // this lane's two output columns
    const int nc1   = nc0 + 16;

    // ---- preload combined B-matrix fragments for both N-tiles -------------
    // B 32x16 f16 layout: col N=l16; elems 0..7 = K=c*32+lhalf*8..,
    //                               elems 8..15 = K=c*32+16+lhalf*8..
    v16h bf0[NCHUNK], bf1[NCHUNK];
#pragma unroll
    for (int c = 0; c < NCHUNK; ++c) {
        const int kr0 = c * 32 + lhalf * 8;
#pragma unroll
        for (int i = 0; i < 16; ++i) {
            const int k = kr0 + i + (i >= 8 ? 8 : 0);
            const float wa = (k < NH) ? Wrec[k * NH + nc0] : Win[(k - NH) * NH + nc0];
            const float wb = (k < NH) ? Wrec[k * NH + nc1] : Win[(k - NH) * NH + nc1];
            bf0[c][i] = (_Float16)wa;
            bf1[c][i] = (_Float16)wb;
        }
    }
    const float br0 = brec[nc0], br1 = brec[nc1];

    // ---- init: r2[0] = relu(x0), u2[0] = u_0, x state + noise[0] in regs --
    for (int e = tid; e < 16 * NH; e += 256) {
        const int m = e >> 8, n = e & 255;
        const float v = x0[(b0 + m) * NH + n];
        r2[0][m * NH + n] = (_Float16)(v > 0.f ? v : 0.f);
    }
    for (int e = tid; e < 16 * NIN; e += 256) {
        u2[0][e] = (_Float16)inputs[(long)b0 * NIN + e];   // rows contiguous
    }
    float xr0[8], xr1[8], nr0[8], nr1[8];
#pragma unroll
    for (int v = 0; v < 8; ++v) {
        const int m = v + lhalf * 8;
        xr0[v] = x0[(b0 + m) * NH + nc0];
        xr1[v] = x0[(b0 + m) * NH + nc1];
        nr0[v] = __builtin_nontemporal_load(&noise[(long)(b0 + m) * NH + nc0]);
        nr1[v] = __builtin_nontemporal_load(&noise[(long)(b0 + m) * NH + nc1]);
    }
    __syncthreads();

    for (int t = 0; t < T_STEPS; ++t) {
        const int p = t & 1, q = p ^ 1;

        // (a) issue next-step input loads early (consumed after the GEMM)
        float ur[4];
        if (t + 1 < T_STEPS) {
            const long ub = (long)(t + 1) * BATCH * NIN + (long)b0 * NIN;
#pragma unroll
            for (int s = 0; s < 4; ++s) ur[s] = inputs[ub + tid + s * 256];
        }

        // (b) GEMM: [r | u] @ [Wrec ; Win], two 16x16 tiles per wave, K=320
        v8f acc0 = {}, acc1 = {};
        const _Float16* rb = r2[p];
        const _Float16* ub = u2[p];
#pragma unroll
        for (int c = 0; c < NCHUNK; ++c) {
            // A 16x32 f16 layout: row M=l16; elems 0..7 = K=c*32+lhalf*8..,
            //                                elems 8..15 = K=c*32+16+lhalf*8..
            const _Float16* ap = (c < 8)
                ? &rb[l16 * NH  + c * 32 + lhalf * 8]
                : &ub[l16 * NIN + (c - 8) * 32 + lhalf * 8];
            const v8h lo = *(const v8h*)ap;         // ds_load_b128
            const v8h hi = *(const v8h*)(ap + 16);  // ds_load_b128
            v16h a;
#pragma unroll
            for (int i = 0; i < 8; ++i) { a[i] = lo[i]; a[i + 8] = hi[i]; }
            acc0 = __builtin_amdgcn_wmma_f32_16x16x32_f16(
                false, a, false, bf0[c], (short)0, acc0, false, false);
            acc1 = __builtin_amdgcn_wmma_f32_16x16x32_f16(
                false, a, false, bf1[c], (short)0, acc1, false, false);
        }

        // (c) stage u[t+1] into the other buffer (GEMM above read u2[p])
        if (t + 1 < T_STEPS) {
            _Float16* ud = u2[q];
#pragma unroll
            for (int s = 0; s < 4; ++s) ud[tid + s * 256] = (_Float16)ur[s];
        }

        // (d) epilogue: leak + noise + bias; state stays in registers.
        // C/D layout: lane col=l16, VGPR v -> row m = v + 8*lhalf.
        const long ob = (long)t * BATCH * NH + (long)b0 * NH;
        _Float16* rd = r2[q];
#pragma unroll
        for (int v = 0; v < 8; ++v) {
            const int m = v + lhalf * 8;
            const float xn0 = (1.f - ALPHA) * xr0[v] + ALPHA * (acc0[v] + br0) + nr0[v];
            const float xn1 = (1.f - ALPHA) * xr1[v] + ALPHA * (acc1[v] + br1) + nr1[v];
            xr0[v] = xn0;
            xr1[v] = xn1;
            __builtin_nontemporal_store(xn0, &x_out[ob + m * NH + nc0]);
            __builtin_nontemporal_store(xn1, &x_out[ob + m * NH + nc1]);
            rd[m * NH + nc0] = (_Float16)(xn0 > 0.f ? xn0 : 0.f);
            rd[m * NH + nc1] = (_Float16)(xn1 > 0.f ? xn1 : 0.f);
        }

        // (e) issue noise[t+1] loads; they complete during next step's GEMM
        if (t + 1 < T_STEPS) {
            const long nb = ob + (long)BATCH * NH;
#pragma unroll
            for (int v = 0; v < 8; ++v) {
                const int m = v + lhalf * 8;
                nr0[v] = __builtin_nontemporal_load(&noise[nb + m * NH + nc0]);
                nr1[v] = __builtin_nontemporal_load(&noise[nb + m * NH + nc1]);
            }
        }

        __syncthreads();   // single barrier: r2[q]/u2[q] writes -> next GEMM reads
    }
}

// ---------------------------------------------------------------------------
// Readout: z = softmax(relu(x) @ Wout + bout).  Fully parallel over T*B rows.
// 8 waves/block, each wave computes a 16x32 tile (2 N-tiles, 8 k-chunks).
// ---------------------------------------------------------------------------
__global__ __launch_bounds__(256) void readout_kernel(
    const float* __restrict__ x,      // (T*B, NH)
    const float* __restrict__ Wout,   // (NH, NOUT)
    const float* __restrict__ bout,   // (NOUT)
    float* __restrict__ z)            // (T*B, NOUT)
{
    __shared__ _Float16 wT[NOUT * NH];        // transposed Wout: wT[n][k], 16 KB
    __shared__ float    bo[NOUT];
    __shared__ float    blk[8][16 * NOUT];    // per-wave 16x32 f32 scratch

    const int tid   = threadIdx.x;
    const int lane  = tid & 31;
    const int wave  = tid >> 5;
    const int lhalf = lane >> 4;
    const int l16   = lane & 15;

    for (int e = tid; e < NH * NOUT; e += 256) {
        const int k = e >> 5, n = e & 31;
        wT[n * NH + k] = (_Float16)Wout[k * NOUT + n];
    }
    if (tid < NOUT) bo[tid] = bout[tid];
    __syncthreads();

    const long row0 = (long)blockIdx.x * 128 + wave * 16;

    v8f acc0 = {}, acc1 = {};
#pragma unroll
    for (int c = 0; c < 8; ++c) {
        // A fragment: relu(x) rows, 16 contiguous k per lane (same k order as B)
        const int ka = c * 32 + lhalf * 16;
        const float* ap = &x[(row0 + l16) * NH + ka];
        v16h a;
#pragma unroll
        for (int i = 0; i < 8; ++i) {
            const float v0 = __builtin_nontemporal_load(ap + i);
            const float v1 = __builtin_nontemporal_load(ap + i + 8);
            a[i]     = (_Float16)(v0 > 0.f ? v0 : 0.f);
            a[i + 8] = (_Float16)(v1 > 0.f ? v1 : 0.f);
        }
        // B fragments: 16 contiguous k per lane from transposed Wout in LDS
        const int kb = c * 32 + lhalf * 16;
        const v16h b0v = *(const v16h*)&wT[l16 * NH + kb];
        const v16h b1v = *(const v16h*)&wT[(16 + l16) * NH + kb];
        acc0 = __builtin_amdgcn_wmma_f32_16x16x32_f16(false, a, false, b0v,
                                                      (short)0, acc0, false, false);
        acc1 = __builtin_amdgcn_wmma_f32_16x16x32_f16(false, a, false, b1v,
                                                      (short)0, acc1, false, false);
    }

#pragma unroll
    for (int v = 0; v < 8; ++v) {
        const int m = v + lhalf * 8;
        blk[wave][m * NOUT + l16]      = acc0[v] + bo[l16];
        blk[wave][m * NOUT + 16 + l16] = acc1[v] + bo[16 + l16];
    }
    __syncthreads();

    // softmax over 32 logits; lanes 0..15 each own one row
    if (lane < 16) {
        const float* rowp = &blk[wave][lane * NOUT];
        float mx = rowp[0];
#pragma unroll
        for (int n = 1; n < NOUT; ++n) mx = fmaxf(mx, rowp[n]);
        float ev[NOUT];
        float s = 0.f;
#pragma unroll
        for (int n = 0; n < NOUT; ++n) { ev[n] = __expf(rowp[n] - mx); s += ev[n]; }
        const float inv = 1.f / s;
        float* zp = &z[(row0 + lane) * NOUT];
#pragma unroll
        for (int n = 0; n < NOUT; ++n) __builtin_nontemporal_store(ev[n] * inv, zp + n);
    }
}

extern "C" void kernel_launch(void* const* d_in, const int* in_sizes, int n_in,
                              void* d_out, int out_size, void* d_ws, size_t ws_size,
                              hipStream_t stream) {
    const float* inputs = (const float*)d_in[0];
    const float* noise  = (const float*)d_in[1];
    const float* x0     = (const float*)d_in[2];
    const float* Win    = (const float*)d_in[3];
    const float* Wrec   = (const float*)d_in[4];
    const float* brec   = (const float*)d_in[5];
    const float* Wout   = (const float*)d_in[6];
    const float* bout   = (const float*)d_in[7];

    float* x_out = (float*)d_out;                       // (T,B,N)
    float* z_out = x_out + (long)T_STEPS * BATCH * NH;  // (T,B,NOUT)

    // Sequential recurrence: 16 independent WGs (16 batch rows each), 8 waves/WG.
    rnn_scan_kernel<<<BATCH / 16, 256, 0, stream>>>(
        inputs, noise, x0, Win, Wrec, brec, x_out);

    // Parallel readout over all T*B rows (depends on x_out; same stream).
    readout_kernel<<<(T_STEPS * BATCH) / 128, 256, 0, stream>>>(
        x_out, Wout, bout, z_out);
}
